// DorefaDense_59846074302837
// MI455X (gfx1250) — 8-bit matrix operand order repaired and checked by reading, compiled, not run on hardware
//
#include <hip/hip_runtime.h>
#include <hip/hip_bf16.h>

typedef __attribute__((ext_vector_type(16))) int   v16i;
typedef __attribute__((ext_vector_type(8)))  float v8f;

#define TM 128
#define TN 128
#define TK 128
#define LDSS 144

#if __has_builtin(__builtin_amdgcn_global_load_async_to_lds_b128)
#define HAS_ASYNC_LDS 1
typedef int v4i_ __attribute__((vector_size(16)));
typedef __attribute__((address_space(1))) v4i_ g_v4i;
typedef __attribute__((address_space(3))) v4i_ l_v4i;
#define ASYNC_CP16(gp, lp) \
    __builtin_amdgcn_global_load_async_to_lds_b128((g_v4i*)(gp), (l_v4i*)(lp), 0, 0)
#else
#define HAS_ASYNC_LDS 0
#endif

__device__ inline void wait_async0() {
#if __has_builtin(__builtin_amdgcn_s_wait_asynccnt)
    __builtin_amdgcn_s_wait_asynccnt(0);
#else
    asm volatile("s_wait_asynccnt 0x0" ::: "memory");
#endif
}

__global__ __launch_bounds__(256) void absum_part(const float* __restrict__ W,
                                                  float* __restrict__ part, int n) {
    __shared__ float red[256];
    float s = 0.f;
    for (int i = blockIdx.x * 256 + threadIdx.x; i < n; i += gridDim.x * 256)
        s += fabsf(W[i]);
    red[threadIdx.x] = s;
    __syncthreads();
    for (int o = 128; o > 0; o >>= 1) {
        if (threadIdx.x < o) red[threadIdx.x] += red[threadIdx.x + o];
        __syncthreads();
    }
    if (threadIdx.x == 0) part[blockIdx.x] = red[0];
}

__global__ __launch_bounds__(256) void absum_final(const float* __restrict__ part,
                                                   float* __restrict__ Es) {
    __shared__ float red[256];
    int t = threadIdx.x;
    red[t] = part[t] + part[t + 256] + part[t + 512] + part[t + 768];
    __syncthreads();
    for (int o = 128; o > 0; o >>= 1) {
        if (t < o) red[t] += red[t + o];
        __syncthreads();
    }
    if (t == 0) *Es = red[0];
}

__global__ __launch_bounds__(256) void quant_act(const float* __restrict__ x,
                                                 unsigned int* __restrict__ A8u, int K) {
    const unsigned int lutlo = 0x44403800u;
    const unsigned int luthi = 0x4E4C4A48u;
    const int row = blockIdx.x;
    const float4* xr = (const float4*)(x + (size_t)row * K);
    unsigned int* ar = A8u + (size_t)row * (K >> 2);
    for (int j = threadIdx.x; j < (K >> 2); j += 256) {
        float4 v = xr[j];
        unsigned q0 = (unsigned)(int)rintf(fminf(fabsf(v.x), 1.f) * 7.f);
        unsigned q1 = (unsigned)(int)rintf(fminf(fabsf(v.y), 1.f) * 7.f);
        unsigned q2 = (unsigned)(int)rintf(fminf(fabsf(v.z), 1.f) * 7.f);
        unsigned q3 = (unsigned)(int)rintf(fminf(fabsf(v.w), 1.f) * 7.f);
        unsigned sel = q0 | (q1 << 8) | (q2 << 16) | (q3 << 24);
        ar[j] = __builtin_amdgcn_perm(luthi, lutlo, sel);
    }
}

__global__ void quant_wt(const float* __restrict__ W, unsigned char* __restrict__ Bt,
                         int K, int N) {
    __shared__ unsigned char t[32][33];
    const int k0 = blockIdx.y * 32, n0 = blockIdx.x * 32;
    for (int i = 0; i < 4; ++i) {
        int r = threadIdx.y * 4 + i;
        t[r][threadIdx.x] =
            (W[(size_t)(k0 + r) * N + n0 + threadIdx.x] >= 0.f) ? 0x38 : 0xB8;
    }
    __syncthreads();
    for (int i = 0; i < 4; ++i) {
        int r = threadIdx.y * 4 + i;
        Bt[(size_t)(n0 + r) * K + k0 + threadIdx.x] = t[threadIdx.x][r];
    }
}

__device__ inline v16i load_afrag(const unsigned char* row, int sel8) {
    v16i f;
    const unsigned char* p = row + sel8;
#pragma unroll
    for (int h = 0; h < 2; ++h) {
        uint2 c0 = *(const uint2*)(p + h * 64 + 0);
        uint2 c1 = *(const uint2*)(p + h * 64 + 16);
        uint2 c2 = *(const uint2*)(p + h * 64 + 32);
        uint2 c3 = *(const uint2*)(p + h * 64 + 48);
        f[h * 8 + 0] = (int)c0.x; f[h * 8 + 1] = (int)c0.y;
        f[h * 8 + 2] = (int)c1.x; f[h * 8 + 3] = (int)c1.y;
        f[h * 8 + 4] = (int)c2.x; f[h * 8 + 5] = (int)c2.y;
        f[h * 8 + 6] = (int)c3.x; f[h * 8 + 7] = (int)c3.y;
    }
    return f;
}
__device__ inline v16i load_bfrag(const unsigned char* col, int sel16) {
    v16i f;
    const unsigned char* p = col + sel16;
#pragma unroll
    for (int h = 0; h < 4; ++h) {
        uint4 c = *(const uint4*)(p + h * 32);
        f[h * 4 + 0] = (int)c.x; f[h * 4 + 1] = (int)c.y;
        f[h * 4 + 2] = (int)c.z; f[h * 4 + 3] = (int)c.w;
    }
    return f;
}

__global__ __launch_bounds__(256)
void dorefa_gemm(const unsigned char* __restrict__ A8,
                 const unsigned char* __restrict__ Bt,
                 const float* __restrict__ bias,
                 const float* __restrict__ Es,
                 float* __restrict__ out,
                 int M, int N, int K) {
    __shared__ unsigned char sA[2][TM * LDSS];
    __shared__ unsigned char sB[2][TN * LDSS];

    const int tid   = threadIdx.x;
    const int lane  = tid & 31;
    const int wid   = tid >> 5;
    const int waveM = wid & 1;
    const int waveN = wid >> 1;
    const int m0 = blockIdx.y * TM;
    const int n0 = blockIdx.x * TN;

    int rr[4], oo[4];
#pragma unroll
    for (int t = 0; t < 4; ++t) {
        int c = tid + t * 256;
        rr[t] = c >> 3;
        oo[t] = (c & 7) * 16;
    }

    v8f acc[4][2] = {};

#if HAS_ASYNC_LDS
#pragma unroll
    for (int t = 0; t < 4; ++t) {
        ASYNC_CP16(A8 + (size_t)(m0 + rr[t]) * K + oo[t], &sA[0][rr[t] * LDSS + oo[t]]);
        ASYNC_CP16(Bt + (size_t)(n0 + rr[t]) * K + oo[t], &sB[0][rr[t] * LDSS + oo[t]]);
    }
    wait_async0();
    __syncthreads();
#else
#pragma unroll
    for (int t = 0; t < 4; ++t) {
        uint4 a = *(const uint4*)(A8 + (size_t)(m0 + rr[t]) * K + oo[t]);
        uint4 b = *(const uint4*)(Bt + (size_t)(n0 + rr[t]) * K + oo[t]);
        *(uint4*)(&sA[0][rr[t] * LDSS + oo[t]]) = a;
        *(uint4*)(&sB[0][rr[t] * LDSS + oo[t]]) = b;
    }
    __syncthreads();
#endif

    const int nk = K / TK;
    const int sel8  = (lane & 16) ? 8 : 0;
    const int sel16 = (lane & 16) ? 16 : 0;
    const int lm = lane & 15;

    for (int kk = 0; kk < nk; ++kk) {
        const int cur = kk & 1, nxt = cur ^ 1;

#if HAS_ASYNC_LDS
        if (kk + 1 < nk) {
            const int kb = (kk + 1) * TK;
#pragma unroll
            for (int t = 0; t < 4; ++t) {
                ASYNC_CP16(A8 + (size_t)(m0 + rr[t]) * K + kb + oo[t],
                           &sA[nxt][rr[t] * LDSS + oo[t]]);
                ASYNC_CP16(Bt + (size_t)(n0 + rr[t]) * K + kb + oo[t],
                           &sB[nxt][rr[t] * LDSS + oo[t]]);
            }
        }
#else
        uint4 pa[4], pb[4];
        if (kk + 1 < nk) {
            const int kb = (kk + 1) * TK;
#pragma unroll
            for (int t = 0; t < 4; ++t) {
                pa[t] = *(const uint4*)(A8 + (size_t)(m0 + rr[t]) * K + kb + oo[t]);
                pb[t] = *(const uint4*)(Bt + (size_t)(n0 + rr[t]) * K + kb + oo[t]);
            }
        }
#endif

        v16i bf0 = load_afrag(&sB[cur][(waveN * 32 + lm) * LDSS], sel8);
        v16i bf1 = load_afrag(&sB[cur][(waveN * 32 + 16 + lm) * LDSS], sel8);
#pragma unroll
        for (int mt = 0; mt < 4; ++mt) {
            v16i af = load_afrag(&sA[cur][(waveM * 64 + mt * 16 + lm) * LDSS], sel8);
            acc[mt][0] = __builtin_amdgcn_wmma_f32_16x16x128_fp8_fp8(
                af, bf0, (short)0, acc[mt][0], false, false);
            acc[mt][1] = __builtin_amdgcn_wmma_f32_16x16x128_fp8_fp8(
                af, bf1, (short)0, acc[mt][1], false, false);
        }

        if (kk + 1 < nk) {
#if HAS_ASYNC_LDS
            wait_async0();
#else
#pragma unroll
            for (int t = 0; t < 4; ++t) {
                *(uint4*)(&sA[nxt][rr[t] * LDSS + oo[t]]) = pa[t];
                *(uint4*)(&sB[nxt][rr[t] * LDSS + oo[t]]) = pb[t];
            }
#endif
            __syncthreads();
        }
    }

    const float E = (*Es) * (1.0f / (4096.0f * 4096.0f));
    const float s = E * (1.0f / 7.0f);
    const int mhi = (lane & 16) ? 8 : 0;
#pragma unroll
    for (int mt = 0; mt < 4; ++mt) {
#pragma unroll
        for (int nt = 0; nt < 2; ++nt) {
            const int n = n0 + waveN * 32 + nt * 16 + lm;
            const float bn = bias[n];
            const int mb = m0 + waveM * 64 + mt * 16 + mhi;
            v8f a = acc[mt][nt];
#pragma unroll
            for (int g = 0; g < 8; ++g) {
                const int m = mb + g;
                out[(size_t)m * N + n] = s * a[g] + bn;
            }
        }
    }
}

extern "C" void kernel_launch(void* const* d_in, const int* in_sizes, int n_in,
                              void* d_out, int out_size, void* d_ws, size_t ws_size,
                              hipStream_t stream) {
    const float* x = (const float*)d_in[0];
    const float* W = (const float*)d_in[1];
    const float* b = (const float*)d_in[2];
    float* out = (float*)d_out;

    const int K = 4096, N = 4096;
    const int M = in_sizes[0] / K;

    char* ws = (char*)d_ws;
    float* Es   = (float*)ws;
    float* part = (float*)(ws + 256);
    unsigned char* A8 = (unsigned char*)(ws + 65536);
    unsigned char* Bt = A8 + (size_t)M * K;

    absum_part<<<1024, 256, 0, stream>>>(W, part, K * N);
    absum_final<<<1, 256, 0, stream>>>(part, Es);
    quant_act<<<M, 256, 0, stream>>>(x, (unsigned int*)A8, K);
    quant_wt<<<dim3(N / 32, K / 32), dim3(32, 8), 0, stream>>>(W, Bt, K, N);
    dorefa_gemm<<<dim3(N / TN, M / TM), 256, 0, stream>>>(A8, Bt, b, Es, out, M, N, K);
}
